// FeRALinear_40776419508270
// MI455X (gfx1250) — matrix operand order repaired and checked by reading, not compiled, not run on hardware
//
#include <hip/hip_runtime.h>
#include <stddef.h>
#include <stdint.h>

// ---------------- problem constants (from reference) ----------------
#define D_IN   4096
#define D_OUT  4096
#define MTOT   8192      // B*S = 4*2048
#define NEXP   8
#define RANK   4
#define JDIM   32        // NEXP*RANK  (rank-32 fused LoRA update)
#define LORA_SCALE 2.0f  // 8.0/4
#define KTILES (D_IN / 32)     // 128
#define MTILES (MTOT / 16)     // 512
#define NTILES (D_OUT / 16)    // 256

// ---------------- types ----------------
typedef __attribute__((ext_vector_type(16))) __bf16 v16bf;
typedef __attribute__((ext_vector_type(8)))  float  v8f;
typedef __attribute__((ext_vector_type(4)))  float  f32x4;
typedef __attribute__((ext_vector_type(4)))  unsigned int u32x4;

union Frag16 { v16bf v; u32x4 q[2]; __bf16 e[16]; };
union Pack8  { u32x4 q; __bf16 e[8]; };

__device__ __forceinline__ void bf16_split(float x, __bf16& h, __bf16& l) {
    h = (__bf16)x;
    l = (__bf16)(x - (float)h);
}

__device__ __forceinline__ v8f wmma_bf16(v16bf a, v16bf b, v8f c) {
    return __builtin_amdgcn_wmma_f32_16x16x32_bf16(false, a, false, b,
                                                   (short)0, c, false, false);
}

// ---- gfx1250 async global->LDS copy (ASYNCcnt-tracked, bypasses VGPRs) ----
__device__ __forceinline__ void async_b128(uint32_t lds_off, const void* gptr) {
    asm volatile("global_load_async_to_lds_b128 %0, %1, off"
                 :: "v"(lds_off), "v"((uint64_t)(uintptr_t)gptr)
                 : "memory");
}
__device__ __forceinline__ void wait_async0() {
    asm volatile("s_wait_asynccnt 0x0" ::: "memory");
}

// ============================================================================
// Split kernels: fp32 -> bf16 hi/lo workspace arrays laid out in exact WMMA
// fragment order, so every (tile,k-step) block is contiguous and the GEMM's
// LDS fill is a pure async copy.
//
// A-frag order used here (contiguous k, same as B): lanes 0-15 hold k 0-15,
// lanes 16-31 hold k 16-31; M = lane%16.
// Layout: [kt][mtile][lane][16]  (1 KB per fragment)
// ============================================================================
__global__ __launch_bounds__(256)
void fera_split_x(const float* __restrict__ X,
                  __bf16* __restrict__ Xh, __bf16* __restrict__ Xl)
{
    const int g  = blockIdx.x * 256 + threadIdx.x;   // one per 16 elements
    const int m  = g >> 8;
    const int k0 = (g & 255) << 4;
    const float* src = X + (size_t)m * D_IN + k0;
    f32x4 v0 = ((const f32x4*)src)[0];
    f32x4 v1 = ((const f32x4*)src)[1];
    f32x4 v2 = ((const f32x4*)src)[2];
    f32x4 v3 = ((const f32x4*)src)[3];
    float f[16] = { v0.x,v0.y,v0.z,v0.w, v1.x,v1.y,v1.z,v1.w,
                    v2.x,v2.y,v2.z,v2.w, v3.x,v3.y,v3.z,v3.w };
    Pack8 h0, h1, l0, l1;
#pragma unroll
    for (int i = 0; i < 8; ++i) bf16_split(f[i],     h0.e[i], l0.e[i]);
#pragma unroll
    for (int i = 0; i < 8; ++i) bf16_split(f[8 + i], h1.e[i], l1.e[i]);

    const int r = m & 15, mt = m >> 4, kt = k0 >> 5;
    const int sbase = ((k0 >> 4) & 1) << 4;          // lane offset 0 or 16
    const size_t base = (size_t)(kt * MTILES + mt) * 32;
    const size_t i0 = (base + r + sbase) * 16;       // slots 0-7 : k first 8
    const size_t i1 = i0 + 8;                        // slots 8-15: k second 8
    *(u32x4*)(Xh + i0) = h0.q;   *(u32x4*)(Xh + i1) = h1.q;
    *(u32x4*)(Xl + i0) = l0.q;   *(u32x4*)(Xl + i1) = l1.q;
}

// B-frag order used here (contiguous k, same as A): lanes 0-15 hold k 0-15, lanes 16-31 hold k 16-31,
// N = lane%16.  Layout: [kt][ntile][lane][16]
__global__ __launch_bounds__(256)
void fera_split_w(const float* __restrict__ W,
                  __bf16* __restrict__ Wh, __bf16* __restrict__ Wl)
{
    const int g  = blockIdx.x * 256 + threadIdx.x;
    const int n  = g >> 8;
    const int k0 = (g & 255) << 4;
    const float* src = W + (size_t)n * D_IN + k0;
    f32x4 v0 = ((const f32x4*)src)[0];
    f32x4 v1 = ((const f32x4*)src)[1];
    f32x4 v2 = ((const f32x4*)src)[2];
    f32x4 v3 = ((const f32x4*)src)[3];
    float f[16] = { v0.x,v0.y,v0.z,v0.w, v1.x,v1.y,v1.z,v1.w,
                    v2.x,v2.y,v2.z,v2.w, v3.x,v3.y,v3.z,v3.w };
    Pack8 h0, h1, l0, l1;
#pragma unroll
    for (int i = 0; i < 8; ++i) bf16_split(f[i],     h0.e[i], l0.e[i]);
#pragma unroll
    for (int i = 0; i < 8; ++i) bf16_split(f[8 + i], h1.e[i], l1.e[i]);

    const int kt   = k0 >> 5;
    const int lane = (n & 15) + ((k0 & 16) ? 16 : 0);
    const size_t base = ((size_t)(kt * NTILES + (n >> 4)) * 32 + lane) * 16;
    *(u32x4*)(Wh + base) = h0.q;   *(u32x4*)(Wh + base + 8) = h1.q;
    *(u32x4*)(Wl + base) = l0.q;   *(u32x4*)(Wl + base + 8) = l1.q;
}

// ============================================================================
// Main kernel: out = X @ W^T (bf16x3 WMMA, fp32 accum) + fused rank-32 LoRA.
// 128x128 tile / workgroup, 8 wave32s (4m x 2n), wave owns 32(m) x 64(n).
// LDS double-buffered; fill is GLOBAL_LOAD_ASYNC_TO_LDS_B128 of contiguous
// 8 KB fragment blocks; pipeline: wait(asynccnt 0) -> barrier -> issue next
// batch -> 24 WMMAs on current buffer.
// ============================================================================
__global__ __launch_bounds__(256)
void fera_main_kernel(const __bf16* __restrict__ Xh, const __bf16* __restrict__ Xl,
                      const __bf16* __restrict__ Wh, const __bf16* __restrict__ Wl,
                      const __bf16* __restrict__ Dhi, const __bf16* __restrict__ Dlo,
                      const __bf16* __restrict__ Uthi, const __bf16* __restrict__ Utlo,
                      float* __restrict__ out)
{
    __shared__ __attribute__((aligned(32))) __bf16 sAhi[2][4096];
    __shared__ __attribute__((aligned(32))) __bf16 sAlo[2][4096];
    __shared__ __attribute__((aligned(32))) __bf16 sBhi[2][4096];
    __shared__ __attribute__((aligned(32))) __bf16 sBlo[2][4096];

    const int tid  = threadIdx.x;
    const int lane = tid & 31;
    const int wv   = tid >> 5;
    const int wm   = wv & 3;
    const int wn   = wv >> 2;

    // 8x8 supertile swizzle: ~32 MB L2 working set -> X/W stay L2-resident.
    const int bid   = blockIdx.x;           // 0..2047
    const int st    = bid >> 6;             // 32 supertiles (8 in m x 4 in n)
    const int loc   = bid & 63;
    const int mb    = (st & 7) * 8 + (loc & 7);
    const int nb    = (st >> 3) * 8 + (loc >> 3);
    const int m0    = mb * 128;
    const int n0    = nb * 128;

    const int o = tid * 32;                 // this thread's 32B of each 8KB block
    const uint32_t lAhi[2] = { (uint32_t)(uintptr_t)&sAhi[0][0],
                               (uint32_t)(uintptr_t)&sAhi[1][0] };
    const uint32_t lAlo[2] = { (uint32_t)(uintptr_t)&sAlo[0][0],
                               (uint32_t)(uintptr_t)&sAlo[1][0] };
    const uint32_t lBhi[2] = { (uint32_t)(uintptr_t)&sBhi[0][0],
                               (uint32_t)(uintptr_t)&sBhi[1][0] };
    const uint32_t lBlo[2] = { (uint32_t)(uintptr_t)&sBlo[0][0],
                               (uint32_t)(uintptr_t)&sBlo[1][0] };

    v8f acc[2][4] = {};

    auto stage = [&](int kt, int buf) {
        const size_t aOff = (size_t)(kt * MTILES + mb * 8) * 1024 + o;  // bytes
        const size_t bOff = (size_t)(kt * NTILES + nb * 8) * 1024 + o;
        async_b128(lAhi[buf] + o,      (const char*)Xh + aOff);
        async_b128(lAhi[buf] + o + 16, (const char*)Xh + aOff + 16);
        async_b128(lAlo[buf] + o,      (const char*)Xl + aOff);
        async_b128(lAlo[buf] + o + 16, (const char*)Xl + aOff + 16);
        async_b128(lBhi[buf] + o,      (const char*)Wh + bOff);
        async_b128(lBhi[buf] + o + 16, (const char*)Wh + bOff + 16);
        async_b128(lBlo[buf] + o,      (const char*)Wl + bOff);
        async_b128(lBlo[buf] + o + 16, (const char*)Wl + bOff + 16);
    };

    stage(0, 0);
    for (int kt = 0; kt < KTILES; ++kt) {
        const int buf = kt & 1;
        wait_async0();          // batch for this kt (issued last iter) is done
        __syncthreads();        // all waves' batches done; prev reads finished
        if (kt + 1 < KTILES) stage(kt + 1, buf ^ 1);   // overlaps the WMMAs

        v16bf ahi[2], alo[2], bhi[4], blo[4];
#pragma unroll
        for (int i = 0; i < 2; ++i) {
            const int t = wm * 2 + i;
            ahi[i] = *(const v16bf*)&sAhi[buf][(t * 32 + lane) << 4];
            alo[i] = *(const v16bf*)&sAlo[buf][(t * 32 + lane) << 4];
        }
#pragma unroll
        for (int j = 0; j < 4; ++j) {
            const int t = wn * 4 + j;
            bhi[j] = *(const v16bf*)&sBhi[buf][(t * 32 + lane) << 4];
            blo[j] = *(const v16bf*)&sBlo[buf][(t * 32 + lane) << 4];
        }
#pragma unroll
        for (int i = 0; i < 2; ++i)
#pragma unroll
            for (int j = 0; j < 4; ++j) {
                acc[i][j] = wmma_bf16(ahi[i], bhi[j], acc[i][j]);  // hi*hi
                acc[i][j] = wmma_bf16(ahi[i], blo[j], acc[i][j]);  // hi*lo
                acc[i][j] = wmma_bf16(alo[i], bhi[j], acc[i][j]);  // lo*hi
            }
    }

    // ---------------- fused rank-32 LoRA epilogue ----------------
    const int r    = lane & 15;
    const int half = lane >> 4;
    v16bf dhi[2], dlo[2], uhi[4], ulo[4];
#pragma unroll
    for (int i = 0; i < 2; ++i) {
        const int m = m0 + (wm * 2 + i) * 16 + r;
        const u32x4* ph = (const u32x4*)(Dhi + (size_t)m * JDIM);
        const u32x4* pl = (const u32x4*)(Dlo + (size_t)m * JDIM);
        Frag16 fh, fl;
        fh.q[0] = ph[2 * half]; fh.q[1] = ph[2 * half + 1];
        fl.q[0] = pl[2 * half]; fl.q[1] = pl[2 * half + 1];
        dhi[i] = fh.v; dlo[i] = fl.v;
    }
#pragma unroll
    for (int j = 0; j < 4; ++j) {
        const int n = n0 + (wn * 4 + j) * 16 + r;
        uhi[j] = *(const v16bf*)(Uthi + (size_t)n * JDIM + (half << 4));
        ulo[j] = *(const v16bf*)(Utlo + (size_t)n * JDIM + (half << 4));
    }
#pragma unroll
    for (int i = 0; i < 2; ++i)
#pragma unroll
        for (int j = 0; j < 4; ++j) {
            acc[i][j] = wmma_bf16(dhi[i], uhi[j], acc[i][j]);
            acc[i][j] = wmma_bf16(dhi[i], ulo[j], acc[i][j]);
            acc[i][j] = wmma_bf16(dlo[i], uhi[j], acc[i][j]);
        }

    // ---------------- write back (C: VGPR v -> M=v+8*half, N=lane%16) ------
#pragma unroll
    for (int i = 0; i < 2; ++i) {
        const int mt = m0 + (wm * 2 + i) * 16 + 8 * half;
#pragma unroll
        for (int j = 0; j < 4; ++j) {
            const int n = n0 + (wn * 4 + j) * 16 + r;
            float* o2 = out + (size_t)mt * D_OUT + n;
#pragma unroll
            for (int v = 0; v < 8; ++v)
                o2[(size_t)v * D_OUT] = acc[i][j][v];
        }
    }
}

// ============================================================================
// Down kernel: D[m, e*4+r] = scale*route[b,e] * (X[m,:] . down_w[e,r,:]).
// A-fragments come straight from the pre-split X (32B contiguous per lane);
// down_w (512 KB, L2 resident) is gathered + split on the fly.
// ============================================================================
__global__ __launch_bounds__(256)
void fera_down_kernel(const __bf16* __restrict__ Xh, const __bf16* __restrict__ Xl,
                      const float*  __restrict__ dw,       // [32][4096]
                      const float*  __restrict__ routing,  // [4][8]
                      __bf16* __restrict__ Dhi, __bf16* __restrict__ Dlo)
{
    const int tid   = threadIdx.x;
    const int lane  = tid & 31;
    const int wv    = tid >> 5;
    const int mbase = blockIdx.x * 128 + wv * 16;
    const int mt    = mbase >> 4;
    const int r     = lane & 15;
    const int half  = lane >> 4;

    v8f acc[2] = {};

    for (int kt = 0; kt < KTILES; ++kt) {
        const size_t fo = (size_t)(kt * MTILES + mt) * 512 + lane * 16;
        v16bf ah = *(const v16bf*)(Xh + fo);
        v16bf al = *(const v16bf*)(Xl + fo);

        const int k0 = kt * 32;
        Frag16 bh[2], bl[2];
#pragma unroll
        for (int jt = 0; jt < 2; ++jt) {
            const float* p = dw + (size_t)(jt * 16 + r) * D_IN + k0 + half * 16;
            f32x4 v0 = ((const f32x4*)p)[0];
            f32x4 v1 = ((const f32x4*)p)[1];
            f32x4 v2 = ((const f32x4*)p)[2];
            f32x4 v3 = ((const f32x4*)p)[3];
            float fb[16] = { v0.x,v0.y,v0.z,v0.w, v1.x,v1.y,v1.z,v1.w,
                             v2.x,v2.y,v2.z,v2.w, v3.x,v3.y,v3.z,v3.w };
#pragma unroll
            for (int i = 0; i < 16; ++i) bf16_split(fb[i], bh[jt].e[i], bl[jt].e[i]);
        }
#pragma unroll
        for (int jt = 0; jt < 2; ++jt) {
            acc[jt] = wmma_bf16(ah, bh[jt].v, acc[jt]);
            acc[jt] = wmma_bf16(ah, bl[jt].v, acc[jt]);
            acc[jt] = wmma_bf16(al, bh[jt].v, acc[jt]);
        }
    }

    const int b = mbase >> 11;  // S = 2048; tiles never straddle a batch
#pragma unroll
    for (int jt = 0; jt < 2; ++jt) {
        const int j = jt * 16 + r;
        const float rw = routing[b * NEXP + (j >> 2)] * LORA_SCALE;
#pragma unroll
        for (int v = 0; v < 8; ++v) {
            const int m = mbase + v + 8 * half;
            __bf16 h, l; bf16_split(acc[jt][v] * rw, h, l);
            Dhi[(size_t)m * JDIM + j] = h;
            Dlo[(size_t)m * JDIM + j] = l;
        }
    }
}

// ============================================================================
// Repack up_w [E][D_OUT][R] -> Ut[n][e*4+r] bf16 hi/lo (B-fragment friendly).
// ============================================================================
__global__ __launch_bounds__(256)
void fera_uprepack_kernel(const float* __restrict__ up_w,
                          __bf16* __restrict__ Uthi, __bf16* __restrict__ Utlo)
{
    const int idx = blockIdx.x * 256 + threadIdx.x;   // 0 .. D_OUT*JDIM-1
    const int n  = idx >> 5;
    const int j  = idx & 31;
    const float v = up_w[(size_t)(j >> 2) * D_OUT * RANK + (size_t)n * RANK + (j & 3)];
    __bf16 h, l; bf16_split(v, h, l);
    Uthi[idx] = h;
    Utlo[idx] = l;
}

// ============================================================================
extern "C" void kernel_launch(void* const* d_in, const int* in_sizes, int n_in,
                              void* d_out, int out_size, void* d_ws, size_t ws_size,
                              hipStream_t stream)
{
    (void)in_sizes; (void)n_in; (void)out_size; (void)ws_size;
    const float* x   = (const float*)d_in[0];  // [4,2048,4096]
    const float* rw  = (const float*)d_in[1];  // [4,8]
    const float* bw  = (const float*)d_in[2];  // [4096,4096]
    const float* dwp = (const float*)d_in[3];  // [8,4,4096]
    const float* uwp = (const float*)d_in[4];  // [8,4096,4]
    float* out = (float*)d_out;

    // workspace layout (~194 MB)
    char* ws = (char*)d_ws;
    const size_t szX = (size_t)MTOT * D_IN * 2;    // 64 MB per array
    const size_t szW = (size_t)D_OUT * D_IN * 2;   // 32 MB per array
    __bf16* Xh = (__bf16*)(ws);
    __bf16* Xl = (__bf16*)(ws + szX);
    __bf16* Wh = (__bf16*)(ws + 2 * szX);
    __bf16* Wl = (__bf16*)(ws + 2 * szX + szW);
    char*   t  = ws + 2 * szX + 2 * szW;
    __bf16* Dhi  = (__bf16*)(t);
    __bf16* Dlo  = (__bf16*)(t + (size_t)MTOT * JDIM * 2);
    __bf16* Uthi = (__bf16*)(t + (size_t)MTOT * JDIM * 4);
    __bf16* Utlo = (__bf16*)(t + (size_t)MTOT * JDIM * 4 + (size_t)D_OUT * JDIM * 2);

    fera_split_x<<<dim3((MTOT * (D_IN / 16)) / 256), 256, 0, stream>>>(x, Xh, Xl);
    fera_split_w<<<dim3((D_OUT * (D_IN / 16)) / 256), 256, 0, stream>>>(bw, Wh, Wl);
    fera_uprepack_kernel<<<dim3((D_OUT * JDIM) / 256), 256, 0, stream>>>(uwp, Uthi, Utlo);
    fera_down_kernel<<<dim3(MTOT / 128), 256, 0, stream>>>(x /*unused*/ ? Xh : Xh, Xl, dwp, rw, Dhi, Dlo);
    fera_main_kernel<<<dim3(2048), 256, 0, stream>>>(Xh, Xl, Wh, Wl,
                                                     Dhi, Dlo, Uthi, Utlo, out);
}
